// DeRA_45389214384191
// MI455X (gfx1250) — matrix operand order repaired and checked by reading, not compiled, not run on hardware
//
#include <hip/hip_runtime.h>
#include <hip/hip_bf16.h>

// ---------------- problem constants (from reference) ----------------
#define DIM   3072
#define RANK  192
#define NH    4
#define HD    48            // head dim
#define TT    8
#define GHh   30
#define GWw   52
#define SEQ   (GHh*GWw)     // 1560
#define SEQP  1600          // SEQ padded to multiple of 64
#define TOT   (TT*SEQ)      // 12480

typedef __attribute__((ext_vector_type(16))) __bf16 v16bf;
typedef __attribute__((ext_vector_type(8)))  __bf16 v8bf;
typedef __attribute__((ext_vector_type(8)))  float  v8f;

__device__ __forceinline__ float rowmax16(float x) {
#pragma unroll
  for (int m = 1; m < 16; m <<= 1) x = fmaxf(x, __shfl_xor(x, m, 32));
  return x;
}

// =====================================================================
// One-shot f32 -> bf16 conversion of all six weight matrices.
// Arena layout: Wd[RANK*DIM] | Wu[DIM*RANK] | Wq | Wk | Wv | Wo
// =====================================================================
#define SWD (RANK*DIM)
#define SWU (DIM*RANK)
#define SWR (RANK*RANK)
#define WTOT (SWD + SWU + 4*SWR)

__global__ void convert_weights(const float* __restrict__ Wd,
                                const float* __restrict__ Wu,
                                const float* __restrict__ Wq,
                                const float* __restrict__ Wk,
                                const float* __restrict__ Wv,
                                const float* __restrict__ Wo,
                                __bf16* __restrict__ out) {
  const int i = blockIdx.x * blockDim.x + threadIdx.x;
  if (i >= WTOT) return;
  const float* src; int off;
  if (i < SWD)            { src = Wd; off = i; }
  else if (i < SWD + SWU) { src = Wu; off = i - SWD; }
  else {
    const int j = i - SWD - SWU;
    const int w = j / SWR;
    off = j - w * SWR;
    src = (w == 0) ? Wq : (w == 1) ? Wk : (w == 2) ? Wv : Wo;
  }
  out[i] = (__bf16)src[off];
}

// =====================================================================
// GEMM: C[M x N] = A[M x Kd] * W[N x Kd]^T (+ bias[N]).
// Kd, N compile-time so all fragment offsets fold into load immediates.
// One wave computes a 16x64 strip (A fragment reused over 4 B tiles).
// =====================================================================
template <int Kd, int N, typename AT, typename OT>
__global__ void __launch_bounds__(256)
gemm_t4(const AT* __restrict__ A, const __bf16* __restrict__ W,
        const float* __restrict__ bias, OT* __restrict__ C, int M) {
  const int lane = threadIdx.x & 31;
  const int half = lane >> 4;
  const int l16  = lane & 15;
  const int wave = blockIdx.x * (blockDim.x >> 5) + (threadIdx.x >> 5);

  constexpr int ntiles = N / 64;
  const int mtiles = M / 16;
  if (wave >= mtiles * ntiles) return;        // uniform per wave

  const int m0 = (wave / ntiles) * 16;
  const int n0 = (wave % ntiles) * 64;

  v8f acc0 = {}, acc1 = {}, acc2 = {}, acc3 = {};
  const AT*     ap = A + (size_t)(m0 + l16) * Kd + 8 * half;
  const __bf16* wp = W + (size_t)(n0 + l16) * Kd + 8 * half;

  for (int k0 = 0; k0 < Kd; k0 += 32) {
    // ---- A fragment: row m0+l16 ; k = k0 + 8*half + e (+16 for e>=8)
    v16bf a;
    if constexpr (__is_same(AT, float)) {
#pragma unroll
      for (int e = 0; e < 8; ++e) {
        a[e]     = (__bf16)ap[e];
        a[8 + e] = (__bf16)ap[16 + e];
      }
    } else {
      const v8bf alo = *(const v8bf*)(ap);
      const v8bf ahi = *(const v8bf*)(ap + 16);
#pragma unroll
      for (int e = 0; e < 8; ++e) { a[e] = alo[e]; a[8 + e] = ahi[e]; }
    }
    ap += 32;

    // ---- 4 B fragments: n = n0 + 16*t + l16 ; k = k0 + 8*half + e (+16 for e>=8)
#pragma unroll
    for (int t = 0; t < 4; ++t) {
      const v8bf blo = *(const v8bf*)(wp + (size_t)t * 16 * Kd);
      const v8bf bhi = *(const v8bf*)(wp + (size_t)t * 16 * Kd + 16);
      v16bf b;
#pragma unroll
      for (int e = 0; e < 8; ++e) { b[e] = blo[e]; b[8 + e] = bhi[e]; }
      if (t == 0) acc0 = __builtin_amdgcn_wmma_f32_16x16x32_bf16(false, a, false, b, (short)0, acc0, false, false);
      if (t == 1) acc1 = __builtin_amdgcn_wmma_f32_16x16x32_bf16(false, a, false, b, (short)0, acc1, false, false);
      if (t == 2) acc2 = __builtin_amdgcn_wmma_f32_16x16x32_bf16(false, a, false, b, (short)0, acc2, false, false);
      if (t == 3) acc3 = __builtin_amdgcn_wmma_f32_16x16x32_bf16(false, a, false, b, (short)0, acc3, false, false);
    }
    wp += 32;
  }

  // ---- epilogue: D element (v,lane) -> row m0+v+8*half, col n0+16*t+l16
#pragma unroll
  for (int t = 0; t < 4; ++t) {
    const int col = n0 + t * 16 + l16;
    const float bv = bias ? bias[col] : 0.0f;
    v8f acc = (t == 0) ? acc0 : (t == 1) ? acc1 : (t == 2) ? acc2 : acc3;
#pragma unroll
    for (int v = 0; v < 8; ++v) {
      const int row = m0 + v + 8 * half;
      C[(size_t)row * N + col] = (OT)(acc[v] + bv);
    }
  }
}

// =====================================================================
// 2D RoPE.  Reads f32 Q,K (GEMM layout (bt,s,head,d)); writes:
//   Qb  bf16 (bt,s,head,48), pre-scaled by 1/sqrt(48)
//   Kp  bf16 (bt,head,key,64) with features 48..63 AND keys >= SEQ zeroed
// pair j: j<8 identity; 8<=j<16 cos/sin[s/GW][j]; 16<=j<24 cos/sin[s%GW][j]
// =====================================================================
__global__ void rope_qk(const float* __restrict__ Qf, const float* __restrict__ Kf,
                        __bf16* __restrict__ Qb, __bf16* __restrict__ Kp,
                        const float* __restrict__ fcos,
                        const float* __restrict__ fsin) {
  const int total = TT * NH * SEQP * 32;        // padded pair grid
  int idx = blockIdx.x * blockDim.x + threadIdx.x;
  if (idx >= total) return;
  const int j    = idx & 31;   int rest = idx >> 5;
  const int s    = rest % SEQP; rest /= SEQP;
  const int head = rest % NH;
  const int bt   = rest / NH;
  const float sc = 0.144337567297406441f;       // 1/sqrt(48)

  __bf16* kp = Kp + (((size_t)(bt * NH + head) * SEQP) + s) * 64 + 2 * j;
  if (s < SEQ && j < 24) {
    float cv = 1.0f, sv = 0.0f;
    if (j >= 8) {
      const int pos = (j < 16) ? (s / GWw) : (s % GWw);
      cv = fcos[pos * 24 + j];
      sv = fsin[pos * 24 + j];
    }
    const size_t base = ((((size_t)bt * SEQ + s) * NH + head) * HD) + 2 * j;
    const float qa = Qf[base], qb = Qf[base + 1];
    Qb[base]     = (__bf16)((qa * cv - qb * sv) * sc);
    Qb[base + 1] = (__bf16)((qa * sv + qb * cv) * sc);
    const float ka = Kf[base], kb = Kf[base + 1];
    kp[0] = (__bf16)(ka * cv - kb * sv);
    kp[1] = (__bf16)(ka * sv + kb * cv);
  } else {                                      // feature / key padding
    kp[0] = (__bf16)0.0f;
    kp[1] = (__bf16)0.0f;
  }
}

// =====================================================================
// V transpose+augment: (bt,s,head,d) bf16 -> (bt,head,feat,key_padded),
// feat 0..47 = V, feat 48 = 1.0 for valid keys (softmax-l column), rest 0.
// =====================================================================
__global__ void transpose_v(const __bf16* __restrict__ Vb, __bf16* __restrict__ Vt) {
  const int total = TT * NH * 64 * SEQP;
  int idx = blockIdx.x * blockDim.x + threadIdx.x;
  if (idx >= total) return;
  const int s = idx % SEQP; int rest = idx / SEQP;
  const int d = rest % 64;  rest /= 64;
  const int head = rest % NH;
  const int bt   = rest / NH;
  __bf16 v = (__bf16)0.0f;
  if (s < SEQ) {
    if (d < HD)       v = Vb[(((size_t)bt * SEQ + s) * NH + head) * HD + d];
    else if (d == HD) v = (__bf16)1.0f;        // ones column -> row sums
  }
  Vt[idx] = v;
}

// =====================================================================
// Flash attention, one wave per (bt, head, 16-query block).
// 25 identical 64-key chunks, no masks/clamps (padding does the work):
//   - 8 WMMA for S = Q.K^T (4 key tiles x K=64 in two halves)
//   - online max via ds_bpermute reduction (only reduction left)
//   - l accumulated by WMMA through the V ones-column (no sum reduction)
//   - P staged via LDS; reads are 4x ds_load_b128 (rows 16B-aligned)
//   - 8 WMMA for [O|l] += P.V
// =====================================================================
__global__ void __launch_bounds__(32)
attn_fa(const __bf16* __restrict__ Q, const __bf16* __restrict__ Kp,
        const __bf16* __restrict__ Vt, __bf16* __restrict__ O) {
  const int qb   = blockIdx.x;
  const int head = blockIdx.y;
  const int bt   = blockIdx.z;
  const int lane = threadIdx.x & 31;
  const int half = lane >> 4;
  const int l16  = lane & 15;
  const int q0   = qb * 16;

  __shared__ __bf16 Plds[16][72];              // 144B rows: 16B-aligned groups

  // ---- Q fragments (features 0..31 and 32..63; >=48 zero), 1/sqrt(48) folded
  int qrow = q0 + l16; if (qrow >= SEQ) qrow = SEQ - 1;
  const __bf16* qp = Q + (((size_t)bt * SEQ + qrow) * NH + head) * HD;
  const __bf16 bz = (__bf16)0.0f;
  v16bf aq0, aq1;
  {
    const int kb = 8 * half;
    const v8bf q_lo = *(const v8bf*)(qp + kb);
    const v8bf q_md = *(const v8bf*)(qp + kb + 16);
    const v8bf q_hi = *(const v8bf*)(qp + kb + 32);
#pragma unroll
    for (int e = 0; e < 8; ++e) {
      aq0[e] = q_lo[e]; aq0[8 + e] = q_md[e];
      aq1[e] = q_hi[e]; aq1[8 + e] = bz;
    }
  }

  float m[8];
  v8f o0 = {}, o1 = {}, o2 = {}, o3 = {};      // o3 = l (ones column)
#pragma unroll
  for (int v = 0; v < 8; ++v) m[v] = -1e30f;

  const __bf16* kcur = Kp + ((size_t)(bt * NH + head) * SEQP + l16) * 64 + 8 * half;
  const __bf16* vcur = Vt + ((size_t)(bt * NH + head) * 64 + l16) * SEQP + 8 * half;

  for (int c = 0; c < SEQP / 64; ++c) {        // 25 chunks of 64 keys
    // ---------- scores: four 16-key tiles ----------
    v8f s0, s1, s2, s3;
#pragma unroll
    for (int t = 0; t < 4; ++t) {
      const v8bf b0 = *(const v8bf*)(kcur + t * 16 * 64);
      const v8bf b1 = *(const v8bf*)(kcur + t * 16 * 64 + 16);
      const v8bf b2 = *(const v8bf*)(kcur + t * 16 * 64 + 32);
      const v8bf b3 = *(const v8bf*)(kcur + t * 16 * 64 + 48);
      v16bf blo, bhi;
#pragma unroll
      for (int e = 0; e < 8; ++e) {
        blo[e] = b0[e]; blo[8 + e] = b1[e];
        bhi[e] = b2[e]; bhi[8 + e] = b3[e];
      }
      v8f ss = {};
      ss = __builtin_amdgcn_wmma_f32_16x16x32_bf16(false, aq0, false, blo, (short)0, ss, false, false);
      ss = __builtin_amdgcn_wmma_f32_16x16x32_bf16(false, aq1, false, bhi, (short)0, ss, false, false);
      if (t == 0) s0 = ss; if (t == 1) s1 = ss; if (t == 2) s2 = ss; if (t == 3) s3 = ss;
    }

    // ---------- online max + exp (l handled by WMMA ones-column) ----------
    float p0[8], p1[8], p2[8], p3[8];
#pragma unroll
    for (int v = 0; v < 8; ++v) {
      float tm = fmaxf(fmaxf(s0[v], s1[v]), fmaxf(s2[v], s3[v]));
      tm = rowmax16(tm);
      const float mn = fmaxf(m[v], tm);
      const float alpha = __expf(m[v] - mn);
      m[v] = mn;
      p0[v] = __expf(s0[v] - mn);
      p1[v] = __expf(s1[v] - mn);
      p2[v] = __expf(s2[v] - mn);
      p3[v] = __expf(s3[v] - mn);
      o0[v] *= alpha; o1[v] *= alpha; o2[v] *= alpha; o3[v] *= alpha;
    }

    // ---------- stage P via LDS (D-frag -> A-frag relayout) ----------
#pragma unroll
    for (int v = 0; v < 8; ++v) {
      __bf16* pr = &Plds[v + 8 * half][l16];
      pr[0]  = (__bf16)p0[v];
      pr[16] = (__bf16)p1[v];
      pr[32] = (__bf16)p2[v];
      pr[48] = (__bf16)p3[v];
    }
    __syncthreads();
    v16bf ap0, ap1;                            // keys 0..31 / 32..63
    {
      const __bf16* prow = &Plds[l16][8 * half];
      const v8bf pa = *(const v8bf*)(prow);
      const v8bf pb = *(const v8bf*)(prow + 16);
      const v8bf pc = *(const v8bf*)(prow + 32);
      const v8bf pd = *(const v8bf*)(prow + 48);
#pragma unroll
      for (int e = 0; e < 8; ++e) {
        ap0[e] = pa[e]; ap0[8 + e] = pb[e];
        ap1[e] = pc[e]; ap1[8 + e] = pd[e];
      }
    }
    __syncthreads();

    // ---------- [O|l] += P . V  (4 feature tiles x 2 key groups) ----------
#pragma unroll
    for (int f = 0; f < 4; ++f) {
      const v8bf va0 = *(const v8bf*)(vcur + (size_t)f * 16 * SEQP);
      const v8bf va1 = *(const v8bf*)(vcur + (size_t)f * 16 * SEQP + 16);
      const v8bf vb0 = *(const v8bf*)(vcur + (size_t)f * 16 * SEQP + 32);
      const v8bf vb1 = *(const v8bf*)(vcur + (size_t)f * 16 * SEQP + 48);
      v16bf bg0, bg1;
#pragma unroll
      for (int e = 0; e < 8; ++e) {
        bg0[e] = va0[e]; bg0[8 + e] = va1[e];
        bg1[e] = vb0[e]; bg1[8 + e] = vb1[e];
      }
      v8f acc = (f == 0) ? o0 : (f == 1) ? o1 : (f == 2) ? o2 : o3;
      acc = __builtin_amdgcn_wmma_f32_16x16x32_bf16(false, ap0, false, bg0, (short)0, acc, false, false);
      acc = __builtin_amdgcn_wmma_f32_16x16x32_bf16(false, ap1, false, bg1, (short)0, acc, false, false);
      if (f == 0) o0 = acc; if (f == 1) o1 = acc; if (f == 2) o2 = acc; if (f == 3) o3 = acc;
    }
    kcur += 64 * 64;
    vcur += 64;
  }

  // ---------- finalize: l lives in col 48 (lane 16*half of o3) ----------
#pragma unroll
  for (int v = 0; v < 8; ++v) {
    const int row = q0 + v + 8 * half;
    const float lv = __shfl(o3[v], 16 * half, 32);
    if (row < SEQ) {
      const float inv = 1.0f / lv;
      __bf16* op = O + ((size_t)bt * SEQ + row) * RANK + head * HD;
      op[l16]      = (__bf16)(o0[v] * inv);
      op[16 + l16] = (__bf16)(o1[v] * inv);
      op[32 + l16] = (__bf16)(o2[v] * inv);
    }
  }
}

// =====================================================================
extern "C" void kernel_launch(void* const* d_in, const int* in_sizes, int n_in,
                              void* d_out, int out_size, void* d_ws, size_t ws_size,
                              hipStream_t stream) {
  const float* x      = (const float*)d_in[0];
  const float* fcos   = (const float*)d_in[6];
  const float* fsin   = (const float*)d_in[7];
  const float* W_down = (const float*)d_in[8];
  const float* W_up   = (const float*)d_in[9];
  const float* Wq     = (const float*)d_in[10];
  const float* bq     = (const float*)d_in[11];
  const float* Wk     = (const float*)d_in[12];
  const float* bk     = (const float*)d_in[13];
  const float* Wv     = (const float*)d_in[14];
  const float* bv     = (const float*)d_in[15];
  const float* Wo     = (const float*)d_in[16];
  const float* bo     = (const float*)d_in[17];
  float* out = (float*)d_out;

  const size_t NR   = (size_t)TOT * RANK;           // 12480*192
  const size_t KPAD = (size_t)TT * NH * SEQP * 64;  // padded K
  const size_t VPAD = (size_t)TT * NH * 64 * SEQP;  // transposed+augmented V

  // ---- workspace carving ----
  char* p = (char*)d_ws;
  float*  Qf  = (float*)p;   p += NR * sizeof(float);
  float*  Kf  = (float*)p;   p += NR * sizeof(float);
  __bf16* wbf = (__bf16*)p;  p += (size_t)WTOT * sizeof(__bf16);
  __bf16* XSb = (__bf16*)p;  p += NR * sizeof(__bf16);
  __bf16* Qb  = (__bf16*)p;  p += NR * sizeof(__bf16);
  __bf16* Kpd = (__bf16*)p;  p += KPAD * sizeof(__bf16);
  __bf16* Vb  = (__bf16*)p;  p += NR * sizeof(__bf16);
  __bf16* Vt  = (__bf16*)p;  p += VPAD * sizeof(__bf16);
  __bf16* Ob  = (__bf16*)p;  p += NR * sizeof(__bf16);
  __bf16* Zb  = (__bf16*)p;  p += NR * sizeof(__bf16);

  __bf16* Wdb = wbf;
  __bf16* Wub = wbf + SWD;
  __bf16* Wqb = wbf + SWD + SWU;
  __bf16* Wkb = Wqb + SWR;
  __bf16* Wvb = Wkb + SWR;
  __bf16* Wob = Wvb + SWR;

  auto blocksFor = [](int M, int N) {
    return ((M / 16) * (N / 64) + 7) / 8;            // 8 waves per block
  };

  // 0) weights f32 -> bf16
  convert_weights<<<(WTOT + 255) / 256, 256, 0, stream>>>(
      W_down, W_up, Wq, Wk, Wv, Wo, wbf);

  // 1) down-projection: XSb = x @ Wd^T  (f32 A converted on the fly)
  gemm_t4<DIM, RANK, float, __bf16><<<blocksFor(TOT, RANK), 256, 0, stream>>>(
      x, Wdb, nullptr, XSb, TOT);

  // 2) QKV projections (+bias): Q,K f32 (for RoPE), V bf16
  gemm_t4<RANK, RANK, __bf16, float><<<blocksFor(TOT, RANK), 256, 0, stream>>>(
      XSb, Wqb, bq, Qf, TOT);
  gemm_t4<RANK, RANK, __bf16, float><<<blocksFor(TOT, RANK), 256, 0, stream>>>(
      XSb, Wkb, bk, Kf, TOT);
  gemm_t4<RANK, RANK, __bf16, __bf16><<<blocksFor(TOT, RANK), 256, 0, stream>>>(
      XSb, Wvb, bv, Vb, TOT);

  // 3) RoPE (Q scaled; K written into padded (bt,head,key,64) buffer)
  {
    const int total = TT * NH * SEQP * 32;
    rope_qk<<<(total + 255) / 256, 256, 0, stream>>>(Qf, Kf, Qb, Kpd, fcos, fsin);
  }

  // 3b) V transpose+augment to (bt,head,feat64,keyp)
  {
    const int total = TT * NH * 64 * SEQP;
    transpose_v<<<(total + 255) / 256, 256, 0, stream>>>(Vb, Vt);
  }

  // 4) flash attention -> Ob (bf16, 12480 x 192)
  {
    dim3 grid((SEQ + 15) / 16, NH, TT);
    attn_fa<<<grid, 32, 0, stream>>>(Qb, Kpd, Vt, Ob);
  }

  // 5) out-projection: Zb = Ob @ Wo^T + bo   (bf16 out)
  gemm_t4<RANK, RANK, __bf16, __bf16><<<blocksFor(TOT, RANK), 256, 0, stream>>>(
      Ob, Wob, bo, Zb, TOT);

  // 6) up-projection: out = Zb @ Wu^T        (f32 out)
  gemm_t4<RANK, DIM, __bf16, float><<<blocksFor(TOT, DIM), 256, 0, stream>>>(
      Zb, Wub, nullptr, out, TOT);
}
